// EnhancedAutoformerAttention_45518063403675
// MI455X (gfx1250) — matrix operand order repaired and checked by reading, not compiled, not run on hardware
//
#include <hip/hip_runtime.h>
#include <hip/hip_bf16.h>

typedef __attribute__((ext_vector_type(16))) __bf16 v16bf;
typedef __attribute__((ext_vector_type(8)))  float  v8f;
typedef __attribute__((ext_vector_type(4)))  unsigned int v4u;
typedef __attribute__((ext_vector_type(8)))  int  v8i;
typedef __attribute__((ext_vector_type(4)))  int  v4i;

#define Bc  4
#define Sc  2048
#define Dc  512
#define Hc  8
#define DKc 64
#define NEG_INF_F (-1000000000.0f)

union BF16Frag { v16bf v; unsigned short u[16]; };

__device__ __forceinline__ unsigned short f2bf(float f) {
  __bf16 h = (__bf16)f;                       // hardware v_cvt_*_bf16_f32
  return __builtin_bit_cast(unsigned short, h);
}
__device__ __forceinline__ float bf2f(unsigned short u) {
  return (float)__builtin_bit_cast(__bf16, u);
}

// ---------------------------------------------------------------------------
// TDM: 2-D tile (tile_dim0=64 contiguous bf16, tile_dim1=64 rows) from a
// [rows x 64] bf16 tensor into LDS.  D# packed per CDNA5 ISA §8.3/8.4.
// amdgpu-toolchain builtin arity: (v4u, v8i, v4i, v4i, v8i, i32 cpol).
// ---------------------------------------------------------------------------
__device__ __forceinline__ void tdm_load_64x64_bf16(const unsigned short* gsrc,
                                                    unsigned lds_off) {
  unsigned long long ga = (unsigned long long)(size_t)gsrc;
  v4u g0;
  g0[0] = 1u;                                        // count=1, user mode
  g0[1] = lds_off;                                   // lds_addr
  g0[2] = (unsigned)ga;                              // global_addr[31:0]
  g0[3] = (unsigned)((ga >> 32) & 0x1FFFFFFu) | (2u << 30); // addr[56:32]|type=2
  v8i g1;
  g1[0] = (int)(1u << 16);                           // data_size=2B
  g1[1] = (int)(64u << 16);                          // tensor_dim0 = 64
  g1[2] = (int)(2048u << 16);                        // tensor_dim1 = 2048
  g1[3] = (int)(64u << 16);                          // tile_dim0 = 64
  g1[4] = (int)64;                                   // tile_dim1 = 64
  g1[5] = (int)64;                                   // tensor_dim0_stride = 64
  g1[6] = 0;
  g1[7] = 0;
  v4i z4 = {0, 0, 0, 0};
  v8i z8 = {0, 0, 0, 0, 0, 0, 0, 0};
  __builtin_amdgcn_tensor_load_to_lds(g0, g1, z4, z4, z8, 0);
}

// ---------------------------------------------------------------------------
// Kernel 1: Y = X @ W^T + b  (X fp32 [8192,512], W fp32 [512,512])
// W tile stored TRANSPOSED in LDS -> B fragment = 16 strided u16 LDS reads.
// Output bf16 [B,H,S,DK]; addTW=1 adds time_weights (Q path).
// ---------------------------------------------------------------------------
__global__ __launch_bounds__(256) void proj_f32(
    const float* __restrict__ X, const float* __restrict__ W,
    const float* __restrict__ bias, const float* __restrict__ tw,
    unsigned short* __restrict__ Yb, int addTW) {
  __shared__ unsigned short WtT[Dc][16];               // [k][n], 16 KB
  const int tid = threadIdx.x, wave = tid >> 5, lane = tid & 31;
  const int n0 = blockIdx.y * 16;
  const int tileM = blockIdx.x * 128 + wave * 16;

  for (int i = tid; i < 16 * Dc; i += 256) {
    const int n = i >> 9, k = i & 511;                 // global-coalesced read
    WtT[k][n] = f2bf(W[(size_t)(n0 + n) * Dc + k]);
  }
  __syncthreads();

  const int koff = (lane >= 16) ? 8 : 0;
  const float* Xrow = X + (size_t)(tileM + (lane & 15)) * Dc;

  v8f acc;
  const float bval = bias[n0 + (lane & 15)];
  #pragma unroll
  for (int i = 0; i < 8; ++i) acc[i] = bval;

  for (int kk = 0; kk < Dc; kk += 32) {
    BF16Frag a, b;
    const float4 fa = *(const float4*)(Xrow + kk + koff);
    const float4 fb = *(const float4*)(Xrow + kk + koff + 4);
    const float4 fc = *(const float4*)(Xrow + kk + koff + 16);
    const float4 fd = *(const float4*)(Xrow + kk + koff + 20);
    a.u[0]=f2bf(fa.x); a.u[1]=f2bf(fa.y); a.u[2]=f2bf(fa.z); a.u[3]=f2bf(fa.w);
    a.u[4]=f2bf(fb.x); a.u[5]=f2bf(fb.y); a.u[6]=f2bf(fb.z); a.u[7]=f2bf(fb.w);
    a.u[8]=f2bf(fc.x); a.u[9]=f2bf(fc.y); a.u[10]=f2bf(fc.z); a.u[11]=f2bf(fc.w);
    a.u[12]=f2bf(fd.x); a.u[13]=f2bf(fd.y); a.u[14]=f2bf(fd.z); a.u[15]=f2bf(fd.w);
    #pragma unroll
    for (int s = 0; s < 16; ++s)   // lane&15 = N, element = K (table order)
      b.u[s] = WtT[kk + koff + (s < 8 ? s : s + 8)][lane & 15];
    acc = __builtin_amdgcn_wmma_f32_16x16x32_bf16(false, a.v, false, b.v,
                                                  (short)0, acc, false, false);
  }

  const int n = n0 + (lane & 15), h = n >> 6, dk = n & 63;
  const float tval = addTW ? tw[h * DKc + dk] : 0.0f;
  #pragma unroll
  for (int r = 0; r < 8; ++r) {
    const int mrow = tileM + r + 8 * (lane >> 4);
    const int bidx = mrow >> 11, sidx = mrow & 2047;
    Yb[((size_t)(bidx * Hc + h) * Sc + sidx) * DKc + dk] = f2bf(acc[r] + tval);
  }
}

// ---------------------------------------------------------------------------
// Kernel 2: gate[b,h,s] = sigmoid(q . gate_w + gate_b)
// ---------------------------------------------------------------------------
__global__ __launch_bounds__(256) void gate_kernel(
    const unsigned short* __restrict__ qb, const float* __restrict__ gw,
    const float* __restrict__ gb, float* __restrict__ gate) {
  const int row = blockIdx.x * blockDim.x + threadIdx.x;
  const unsigned short* q = qb + (size_t)row * DKc;
  float s = 0.0f;
  #pragma unroll
  for (int i = 0; i < DKc; ++i) s += bf2f(q[i]) * gw[i];
  gate[row] = 1.0f / (1.0f + __expf(-(s + gb[0])));
}

// ---------------------------------------------------------------------------
// Kernel 3: flash attention, transposed score layout (S^T = K x Q^T):
//  - K tile is the A operand (contiguous LDS b128 reads)
//  - Q becomes a B fragment, built ONCE per kernel
//  - mask loads vectorize to int4; softmax row = 1 shfl_xor
// K/V chunks staged by the Tensor Data Mover, double-buffered;
// overlap via in-order TENSORcnt partial waits.
// ---------------------------------------------------------------------------
__global__ __launch_bounds__(256) void attn_kernel(
    const unsigned short* __restrict__ qb, const unsigned short* __restrict__ kb,
    const unsigned short* __restrict__ vb, const int* __restrict__ mask,
    const float* __restrict__ gate, const float* __restrict__ temp,
    unsigned short* __restrict__ ctxb) {
  __shared__ unsigned short Kt[2][64][DKc];            // 16 KB (double buffer)
  __shared__ unsigned short Vt[2][64][DKc];            // 16 KB
  __shared__ unsigned short Pw[8][16][64];             // 16 KB (per-wave P)

  const int tid = threadIdx.x, wave = tid >> 5, lane = tid & 31;
  const int bh = blockIdx.y, b = bh >> 3, h = bh & 7;
  const int qbase = blockIdx.x * 128 + wave * 16;
  const int koff  = (lane >= 16) ? 8 : 0;
  const float invT = 1.0f / temp[0];

  // Q as B fragments (WMMA-K = d): lane&15 = q-row, element j = d (one-time cost)
  BF16Frag bQ0, bQ1;
  #pragma unroll
  for (int j = 0; j < 16; ++j) {
    const unsigned short* qp = qb + ((size_t)bh * Sc + qbase + (lane & 15)) * DKc;
    const int kd = koff + (j < 8 ? j : j + 8);
    bQ0.u[j] = qp[kd];
    bQ1.u[j] = qp[32 + kd];
  }
  const float gq = gate[(size_t)bh * Sc + qbase + (lane & 15)]; // per-lane q

  float mrun = -3.0e38f, lrun = 0.0f;                  // stats for q = lane&15
  v8f o[4];
  #pragma unroll
  for (int t = 0; t < 4; ++t)
    #pragma unroll
    for (int i = 0; i < 8; ++i) o[t][i] = 0.0f;

  const unsigned short* kbh = kb + (size_t)bh * Sc * DKc;
  const unsigned short* vbh = vb + (size_t)bh * Sc * DKc;

  // prologue: TDM chunk 0 into buffer 0 (wave 0 only; EXEC-independent DMA)
  if (wave == 0) {
    tdm_load_64x64_bf16(kbh, (unsigned)(size_t)&Kt[0][0][0]);
    tdm_load_64x64_bf16(vbh, (unsigned)(size_t)&Vt[0][0][0]);
  }

  for (int kc = 0; kc < Sc; kc += 64) {
    const int cur = (kc >> 6) & 1, nxt = cur ^ 1;
    __syncthreads();                                   // nxt buffer now free
    if (wave == 0) {
      if (kc + 64 < Sc) {
        tdm_load_64x64_bf16(kbh + (size_t)(kc + 64) * DKc,
                            (unsigned)(size_t)&Kt[nxt][0][0]);
        tdm_load_64x64_bf16(vbh + (size_t)(kc + 64) * DKc,
                            (unsigned)(size_t)&Vt[nxt][0][0]);
        __builtin_amdgcn_s_wait_tensorcnt(2);          // cur pair done
      } else {
        __builtin_amdgcn_s_wait_tensorcnt(0);
      }
    }
    // prefetch next chunk's mask stream (the largest HBM stream: 67 MB)
    if (kc + 64 < Sc) {
      const int pr = blockIdx.x * 128 + (tid >> 1);
      __builtin_prefetch(&mask[((size_t)b * Sc + pr) * Sc + kc + 64 + (tid & 1) * 32], 0, 0);
    }
    __syncthreads();                                   // cur data visible

    // ---- S^T tiles: 4 tiles of 16 keys; A = K tile, B = Q fragment ----
    float st[4][8];
    float cmax = -3.0e38f;
    #pragma unroll
    for (int t = 0; t < 4; ++t) {
      BF16Frag aK0, aK1;                               // lane&15 = key row
      const unsigned short* kr = &Kt[cur][t * 16 + (lane & 15)][0];
      *(uint4*)&aK0.u[0] = *(const uint4*)(kr + koff);
      *(uint4*)&aK0.u[8] = *(const uint4*)(kr + koff + 16);
      *(uint4*)&aK1.u[0] = *(const uint4*)(kr + 32 + koff);
      *(uint4*)&aK1.u[8] = *(const uint4*)(kr + 32 + koff + 16);
      v8f c;
      #pragma unroll
      for (int i = 0; i < 8; ++i) c[i] = 0.0f;
      c = __builtin_amdgcn_wmma_f32_16x16x32_bf16(false, aK0.v, false, bQ0.v,
                                                  (short)0, c, false, false);
      c = __builtin_amdgcn_wmma_f32_16x16x32_bf16(false, aK1.v, false, bQ1.v,
                                                  (short)0, c, false, false);
      // C element (r,l): key = kc + t*16 + r + 8*(l>>4), q = qbase + (l&15)
      const int q = qbase + (lane & 15);
      const int keyb = kc + t * 16 + 8 * (lane >> 4);  // contiguous over r
      const int4 m0 = *(const int4*)&mask[((size_t)b * Sc + q) * Sc + keyb];
      const int4 m1 = *(const int4*)&mask[((size_t)b * Sc + q) * Sc + keyb + 4];
      const int mv[8] = {m0.x, m0.y, m0.z, m0.w, m1.x, m1.y, m1.z, m1.w};
      #pragma unroll
      for (int r = 0; r < 8; ++r) {
        float v = (mv[r] == 0) ? NEG_INF_F : c[r] * invT;
        v *= gq;
        st[t][r] = v;
        cmax = fmaxf(cmax, v);
      }
    }
    // ---- online softmax (per q = lane&15; partner lane holds other keys) --
    cmax = fmaxf(cmax, __shfl_xor(cmax, 16, 32));
    const float mnew = fmaxf(mrun, cmax);
    const float scl  = __expf(mrun - mnew);
    float rs = 0.0f;
    #pragma unroll
    for (int t = 0; t < 4; ++t) {
      unsigned short pk[8];
      #pragma unroll
      for (int r = 0; r < 8; ++r) {
        const float p = __expf(st[t][r] - mnew);
        rs += p;
        pk[r] = f2bf(p);
      }
      // one b128 store: keys contiguous over r
      *(uint4*)&Pw[wave][lane & 15][t * 16 + 8 * (lane >> 4)] = *(const uint4*)pk;
    }
    rs += __shfl_xor(rs, 16, 32);
    lrun = lrun * scl + rs;
    mrun = mnew;

    asm volatile("s_wait_dscnt 0x0" ::: "memory");     // P RAW within wave

    // rescale accumulators: o rows are q = r + 8*(l>>4) -> redistribute scl
    float sclr[8];
    #pragma unroll
    for (int r = 0; r < 8; ++r) sclr[r] = __shfl(scl, r + 8 * (lane >> 4), 32);
    #pragma unroll
    for (int t = 0; t < 4; ++t)
      #pragma unroll
      for (int r = 0; r < 8; ++r) o[t][r] *= sclr[r];

    // ---- P @ V ----
    BF16Frag aP0, aP1;                                 // A: M=q, K=key
    const unsigned short* pr2 = &Pw[wave][lane & 15][0];
    *(uint4*)&aP0.u[0] = *(const uint4*)(pr2 + koff);
    *(uint4*)&aP0.u[8] = *(const uint4*)(pr2 + koff + 16);
    *(uint4*)&aP1.u[0] = *(const uint4*)(pr2 + 32 + koff);
    *(uint4*)&aP1.u[8] = *(const uint4*)(pr2 + 32 + koff + 16);
    #pragma unroll
    for (int t = 0; t < 4; ++t) {
      BF16Frag bv0, bv1;                               // B: lane&15 = d, elem = key
      #pragma unroll
      for (int s = 0; s < 16; ++s) {
        const int ky = koff + (s < 8 ? s : s + 8);
        bv0.u[s] = Vt[cur][ky][t * 16 + (lane & 15)];
        bv1.u[s] = Vt[cur][32 + ky][t * 16 + (lane & 15)];
      }
      o[t] = __builtin_amdgcn_wmma_f32_16x16x32_bf16(false, aP0.v, false, bv0.v,
                                                     (short)0, o[t], false, false);
      o[t] = __builtin_amdgcn_wmma_f32_16x16x32_bf16(false, aP1.v, false, bv1.v,
                                                     (short)0, o[t], false, false);
    }
  }

  // normalize (redistribute lrun to C-row layout) + write ctx bf16 [B,S,D]
  float lr[8];
  #pragma unroll
  for (int r = 0; r < 8; ++r) lr[r] = __shfl(lrun, r + 8 * (lane >> 4), 32);
  #pragma unroll
  for (int t = 0; t < 4; ++t)
    #pragma unroll
    for (int r = 0; r < 8; ++r) {
      const int qrow = qbase + r + 8 * (lane >> 4);
      const int d = h * DKc + t * 16 + (lane & 15);
      ctxb[((size_t)b * Sc + qrow) * Dc + d] = f2bf(o[t][r] / lr[r]);
    }
}

// ---------------------------------------------------------------------------
// Kernel 4: out = ctx(bf16) @ Wo^T + bo -> fp32 [8192,512]
// ---------------------------------------------------------------------------
__global__ __launch_bounds__(256) void gemm_bf16x(
    const unsigned short* __restrict__ X, const float* __restrict__ W,
    const float* __restrict__ bias, float* __restrict__ Y) {
  __shared__ unsigned short WtT[Dc][16];
  const int tid = threadIdx.x, wave = tid >> 5, lane = tid & 31;
  const int n0 = blockIdx.y * 16;
  const int tileM = blockIdx.x * 128 + wave * 16;

  for (int i = tid; i < 16 * Dc; i += 256) {
    const int n = i >> 9, k = i & 511;
    WtT[k][n] = f2bf(W[(size_t)(n0 + n) * Dc + k]);
  }
  __syncthreads();

  const int koff = (lane >= 16) ? 8 : 0;
  const unsigned short* Xrow = X + (size_t)(tileM + (lane & 15)) * Dc;

  v8f acc;
  const float bval = bias[n0 + (lane & 15)];
  #pragma unroll
  for (int i = 0; i < 8; ++i) acc[i] = bval;

  for (int kk = 0; kk < Dc; kk += 32) {
    BF16Frag a, b;
    *(uint4*)&a.u[0] = *(const uint4*)(Xrow + kk + koff);
    *(uint4*)&a.u[8] = *(const uint4*)(Xrow + kk + koff + 16);
    #pragma unroll
    for (int s = 0; s < 16; ++s)
      b.u[s] = WtT[kk + koff + (s < 8 ? s : s + 8)][lane & 15];
    acc = __builtin_amdgcn_wmma_f32_16x16x32_bf16(false, a.v, false, b.v,
                                                  (short)0, acc, false, false);
  }
  #pragma unroll
  for (int r = 0; r < 8; ++r) {
    const int mrow = tileM + r + 8 * (lane >> 4);
    Y[(size_t)mrow * Dc + n0 + (lane & 15)] = acc[r];
  }
}

// ---------------------------------------------------------------------------
// Kernel 5: row-wise layernorm, one wave per row of 512.
// ---------------------------------------------------------------------------
__global__ __launch_bounds__(256) void ln_kernel(
    const float* __restrict__ X, const float* __restrict__ gam,
    const float* __restrict__ bet, float* __restrict__ Y) {
  const int wave = threadIdx.x >> 5, lane = threadIdx.x & 31;
  const int row = blockIdx.x * 8 + wave;
  const float* xr = X + (size_t)row * Dc;
  float v[16], s = 0.0f, s2 = 0.0f;
  #pragma unroll
  for (int i = 0; i < 16; ++i) {
    v[i] = xr[i * 32 + lane];
    s += v[i]; s2 += v[i] * v[i];
  }
  #pragma unroll
  for (int m = 1; m < 32; m <<= 1) { s += __shfl_xor(s, m, 32); s2 += __shfl_xor(s2, m, 32); }
  const float mu = s * (1.0f / Dc);
  const float var = s2 * (1.0f / Dc) - mu * mu;
  const float rstd = rsqrtf(var + 1e-5f);
  float* yr = Y + (size_t)row * Dc;
  #pragma unroll
  for (int i = 0; i < 16; ++i) {
    const int col = i * 32 + lane;
    yr[col] = (v[i] - mu) * rstd * gam[col] + bet[col];
  }
}

// ---------------------------------------------------------------------------
extern "C" void kernel_launch(void* const* d_in, const int* in_sizes, int n_in,
                              void* d_out, int out_size, void* d_ws, size_t ws_size,
                              hipStream_t stream) {
  const float* Q    = (const float*)d_in[0];
  const float* K    = (const float*)d_in[1];
  const float* V    = (const float*)d_in[2];
  const int*   msk  = (const int*)d_in[3];
  const float* Wq   = (const float*)d_in[4];
  const float* bq   = (const float*)d_in[5];
  const float* Wk   = (const float*)d_in[6];
  const float* bk   = (const float*)d_in[7];
  const float* Wv   = (const float*)d_in[8];
  const float* bv   = (const float*)d_in[9];
  const float* Wo   = (const float*)d_in[10];
  const float* bo   = (const float*)d_in[11];
  const float* temp = (const float*)d_in[12];
  const float* tw   = (const float*)d_in[13];
  const float* gw   = (const float*)d_in[14];
  const float* gb   = (const float*)d_in[15];
  const float* lng  = (const float*)d_in[16];
  const float* lnb  = (const float*)d_in[17];

  char* ws = (char*)d_ws;
  size_t off = 0;
  const size_t nBHS_DK = (size_t)Bc * Hc * Sc * DKc;
  unsigned short* qbuf = (unsigned short*)(ws + off); off += nBHS_DK * 2;
  unsigned short* kbuf = (unsigned short*)(ws + off); off += nBHS_DK * 2;
  unsigned short* vbuf = (unsigned short*)(ws + off); off += nBHS_DK * 2;
  float* gbuf          = (float*)(ws + off);          off += (size_t)Bc * Hc * Sc * 4;
  unsigned short* ctxb = (unsigned short*)(ws + off); off += (size_t)Bc * Sc * Dc * 2;
  float* outf          = (float*)(ws + off);          off += (size_t)Bc * Sc * Dc * 4;
  (void)ws_size; (void)in_sizes; (void)n_in; (void)out_size;

  const dim3 gGemm(64, 32), blk(256);
  proj_f32<<<gGemm, blk, 0, stream>>>(Q, Wq, bq, tw, qbuf, 1);
  proj_f32<<<gGemm, blk, 0, stream>>>(K, Wk, bk, tw, kbuf, 0);
  proj_f32<<<gGemm, blk, 0, stream>>>(V, Wv, bv, tw, vbuf, 0);
  gate_kernel<<<(Bc * Hc * Sc) / 256, blk, 0, stream>>>(qbuf, gw, gb, gbuf);
  attn_kernel<<<dim3(Sc / 128, Bc * Hc), blk, 0, stream>>>(qbuf, kbuf, vbuf, msk,
                                                           gbuf, temp, ctxb);
  gemm_bf16x<<<gGemm, blk, 0, stream>>>(ctxb, Wo, bo, outf);
  ln_kernel<<<(Bc * Sc) / 8, blk, 0, stream>>>(outf, lng, lnb, (float*)d_out);
}
